// SelfAttention2D_37761352467034
// MI455X (gfx1250) — matrix operand order repaired and checked by reading, not compiled, not run on hardware
//
#include <hip/hip_runtime.h>
#include <hip/hip_bf16.h>
#include <stdint.h>

// ---------------------------------------------------------------------------
// SelfAttention2D for MI455X (gfx1250, wave32, WMMA bf16 16x16x32)
// B=16, C=512, H=W=32 (N=1024), heads=8 (d=64), groups=32, eps=1e-5
// ---------------------------------------------------------------------------

#define Bc 16
#define Cc 512
#define Nc 1024
#define NHEAD 8
#define DHEAD 64
#define CPG 16      // channels per group (512/32)

typedef __attribute__((ext_vector_type(16))) __bf16 v16bf;
typedef __attribute__((ext_vector_type(8)))  float  v8f;

union BFrag {        // one WMMA 16x16x32 bf16 operand: 32 bytes / lane
    v16bf    bf;
    uint4    q[2];
    uint32_t u[8];
};

__device__ __forceinline__ uint16_t f2bf(float x) {      // RNE f32 -> bf16
    uint32_t u = __float_as_uint(x);
    return (uint16_t)((u + 0x7FFFu + ((u >> 16) & 1u)) >> 16);
}
__device__ __forceinline__ uint32_t pack2bf(float a, float b) {
    return (uint32_t)f2bf(a) | ((uint32_t)f2bf(b) << 16);
}
__device__ __forceinline__ v8f wmma_bf16(const BFrag& a, const BFrag& b, v8f c) {
    return __builtin_amdgcn_wmma_f32_16x16x32_bf16(false, a.bf, false, b.bf,
                                                   (short)0, c, false, false);
}

#define LOG2E 1.44269504088896340736f

// ---------------------------------------------------------------------------
// Kernel 1: convert weights f32 -> bf16 (qkv_w [1536,512], proj_w [512,512])
// ---------------------------------------------------------------------------
__global__ void cvt_weights_kernel(const float* __restrict__ qkv_w,
                                   const float* __restrict__ proj_w,
                                   uint16_t* __restrict__ qkv_wb,
                                   uint16_t* __restrict__ proj_wb) {
    int i = blockIdx.x * 256 + threadIdx.x;
    const int nq = 3 * Cc * Cc;
    const int np = Cc * Cc;
    if (i < nq) qkv_wb[i] = f2bf(qkv_w[i]);
    if (i < np) proj_wb[i] = f2bf(proj_w[i]);
}

// ---------------------------------------------------------------------------
// Kernel 2: GroupNorm + transpose -> hT[b][n][c] bf16 (n-major) so GEMM
// B-fragments are two contiguous 16B loads per lane.
// ---------------------------------------------------------------------------
__global__ void groupnorm_kernel(const float* __restrict__ x,
                                 const float* __restrict__ gn_w,
                                 const float* __restrict__ gn_b,
                                 uint16_t* __restrict__ hT) {
    int blk = blockIdx.x;            // b*32 + g
    int b = blk >> 5, g = blk & 31;
    int c0 = g * CPG;
    const float* xp = x + ((size_t)b * Cc + c0) * Nc;

    __shared__ float red[256];
    const int NTOT = CPG * Nc;       // 16384
    float s = 0.f, ss = 0.f;
    for (int i = threadIdx.x; i < NTOT; i += 256) {
        float v = xp[i];
        s += v; ss += v * v;
    }
    red[threadIdx.x] = s; __syncthreads();
    for (int o = 128; o > 0; o >>= 1) {
        if ((int)threadIdx.x < o) red[threadIdx.x] += red[threadIdx.x + o];
        __syncthreads();
    }
    float mean = red[0] * (1.0f / NTOT);
    __syncthreads();
    red[threadIdx.x] = ss; __syncthreads();
    for (int o = 128; o > 0; o >>= 1) {
        if ((int)threadIdx.x < o) red[threadIdx.x] += red[threadIdx.x + o];
        __syncthreads();
    }
    float var  = red[0] * (1.0f / NTOT) - mean * mean;
    float rstd = rsqrtf(var + 1e-5f);

    for (int i = threadIdx.x; i < NTOT; i += 256) {
        int cc = i >> 10, n = i & 1023;
        int c = c0 + cc;
        float v = (xp[i] - mean) * rstd * gn_w[c] + gn_b[c];
        hT[((size_t)b * Nc + n) * Cc + c] = f2bf(v);
    }
}

// ---------------------------------------------------------------------------
// Kernel 3: QKV GEMM.  Per batch: W[1536,512] x hT^T.  Wave tile M=64,N=32.
// q is scaled by (1/sqrt(d))*log2(e) so attention runs in the log2 domain.
// Layouts: q,k -> [b,h,n,d]; v -> [b,h,d,n].
// ---------------------------------------------------------------------------
__global__ void qkv_gemm_kernel(const uint16_t* __restrict__ Wb,
                                const uint16_t* __restrict__ hT,
                                const float*   __restrict__ qkv_b,
                                uint16_t* __restrict__ qT,
                                uint16_t* __restrict__ kT,
                                uint16_t* __restrict__ vv) {
    const int lane  = threadIdx.x & 31;
    const int l16   = lane & 15;
    const int sel8  = (lane < 16) ? 0 : 8;
    const int sel16 = (lane < 16) ? 0 : 16;

    int wid = blockIdx.x * 4 + (threadIdx.x >> 5);
    int b  = wid / (24 * 32);
    int r2 = wid % (24 * 32);
    int M0 = (r2 >> 5) * 64;
    int n0 = (r2 & 31) * 32;

    v8f acc[4][2];
    #pragma unroll
    for (int t = 0; t < 4; t++)
        #pragma unroll
        for (int s = 0; s < 2; s++)
            #pragma unroll
            for (int i = 0; i < 8; i++) acc[t][s][i] = 0.f;

    const uint16_t* hTb = hT + (size_t)b * Nc * Cc;
    for (int kb = 0; kb < Cc; kb += 32) {
        BFrag afr[4], bfr[2];
        #pragma unroll
        for (int t = 0; t < 4; t++) {
            const uint16_t* ap = Wb + (size_t)(M0 + 16 * t + l16) * Cc + kb + sel8;
            afr[t].q[0] = *(const uint4*)ap;
            afr[t].q[1] = *(const uint4*)(ap + 16);
        }
        #pragma unroll
        for (int s = 0; s < 2; s++) {
            const uint16_t* bp = hTb + (size_t)(n0 + 16 * s + l16) * Cc + kb + sel8;
            bfr[s].q[0] = *(const uint4*)bp;
            bfr[s].q[1] = *(const uint4*)(bp + 16);
        }
        #pragma unroll
        for (int t = 0; t < 4; t++)
            #pragma unroll
            for (int s = 0; s < 2; s++)
                acc[t][s] = wmma_bf16(afr[t], bfr[s], acc[t][s]);
    }

    int region = M0 / Cc;                 // 0=q, 1=k, 2=v
    int head   = (M0 % Cc) / DHEAD;

    #pragma unroll
    for (int t = 0; t < 4; t++) {
        int obase = M0 + 16 * t + sel8;
        #pragma unroll
        for (int s = 0; s < 2; s++) {
            int n = n0 + 16 * s + l16;
            if (region < 2) {
                float sc = (region == 0) ? (0.125f * LOG2E) : 1.0f;
                uint32_t w0 = pack2bf((acc[t][s][0] + qkv_b[obase + 0]) * sc,
                                      (acc[t][s][1] + qkv_b[obase + 1]) * sc);
                uint32_t w1 = pack2bf((acc[t][s][2] + qkv_b[obase + 2]) * sc,
                                      (acc[t][s][3] + qkv_b[obase + 3]) * sc);
                uint32_t w2 = pack2bf((acc[t][s][4] + qkv_b[obase + 4]) * sc,
                                      (acc[t][s][5] + qkv_b[obase + 5]) * sc);
                uint32_t w3 = pack2bf((acc[t][s][6] + qkv_b[obase + 6]) * sc,
                                      (acc[t][s][7] + qkv_b[obase + 7]) * sc);
                size_t rowoff = ((size_t)(b * NHEAD + head) * Nc + n) * DHEAD
                                + 16 * t + sel8;
                uint16_t* dst = (region == 0 ? qT : kT) + rowoff;
                *(uint4*)dst = make_uint4(w0, w1, w2, w3);
            } else {
                #pragma unroll
                for (int r = 0; r < 8; r++) {
                    int dd = 16 * t + r + sel8;
                    float v = acc[t][s][r] + qkv_b[obase + r];
                    vv[((size_t)(b * NHEAD + head) * DHEAD + dd) * Nc + n] = f2bf(v);
                }
            }
        }
    }
}

// ---------------------------------------------------------------------------
// Kernel 4: flash attention, zero LDS.  One wave owns 16 query rows.
// Computes S^T (A = k rows, B = q cols) so softmax stats are per-lane;
// the S^T C-tile pair re-packs IN REGISTERS into the A-fragment of P for
// the PV WMMA (half-wave row split == A-fragment K pattern).  Online
// softmax in log2 domain (log2(e)/sqrt(d) pre-folded into q).
// ---------------------------------------------------------------------------
__global__ void attn_kernel(const uint16_t* __restrict__ qT,
                            const uint16_t* __restrict__ kT,
                            const uint16_t* __restrict__ vv,
                            uint16_t* __restrict__ aT) {
    const int lane  = threadIdx.x & 31;
    const int l16   = lane & 15;
    const int sel8  = (lane < 16) ? 0 : 8;
    const int sel16 = (lane < 16) ? 0 : 16;

    int wid = blockIdx.x * 8 + (threadIdx.x >> 5);
    int bh  = wid >> 6;                  // b*8 + head, 0..127
    int n_base = (wid & 63) * 16;        // 16-query strip

    const uint16_t* qTb = qT + (size_t)bh * Nc * DHEAD;
    const uint16_t* kTb = kT + (size_t)bh * Nc * DHEAD;
    const uint16_t* vb  = vv + (size_t)bh * DHEAD * Nc;

    // q as B-fragments (column = query, K = dd), held all kernel
    BFrag qf[2];
    {
        const uint16_t* qp = qTb + (size_t)(n_base + l16) * DHEAD;
        #pragma unroll
        for (int kk = 0; kk < 2; kk++) {
            qf[kk].q[0] = *(const uint4*)(qp + kk * 32 + sel8);
            qf[kk].q[1] = *(const uint4*)(qp + kk * 32 + sel8 + 16);
        }
    }

    v8f oacc[4];                          // out rows = queries, cols = dd
    #pragma unroll
    for (int s = 0; s < 4; s++)
        #pragma unroll
        for (int i = 0; i < 8; i++) oacc[s][i] = 0.f;

    float mrun = -3.0e38f;                // running max (log2 domain), per query lane
    float lrun = 0.f;                     // running sum (this half's keys)

    for (int j0 = 0; j0 < Nc; j0 += 32) {
        // ---- S^T tiles: rows = keys (2 x 16), cols = queries ----
        v8f st[2];
        #pragma unroll
        for (int u = 0; u < 2; u++) {
            const uint16_t* kp = kTb + (size_t)(j0 + 16 * u + l16) * DHEAD;
            BFrag ka0, ka1;
            ka0.q[0] = *(const uint4*)(kp + sel8);
            ka0.q[1] = *(const uint4*)(kp + sel8 + 16);
            ka1.q[0] = *(const uint4*)(kp + 32 + sel8);
            ka1.q[1] = *(const uint4*)(kp + 32 + sel8 + 16);
            v8f c = {0.f, 0.f, 0.f, 0.f, 0.f, 0.f, 0.f, 0.f};
            c = wmma_bf16(ka0, qf[0], c);
            c = wmma_bf16(ka1, qf[1], c);
            st[u] = c;
        }

        // ---- per-query block max: in-lane over 16 keys + one cross-half ----
        float bm = fmaxf(st[0][0], st[1][0]);
        #pragma unroll
        for (int e = 1; e < 8; e++) bm = fmaxf(bm, fmaxf(st[0][e], st[1][e]));
        bm = fmaxf(bm, __shfl_xor(bm, 16, 32));
        float mnew  = fmaxf(mrun, bm);
        float alpha = exp2f(mrun - mnew);

        // ---- P = exp2(st - mnew): packs directly into an A-fragment ----
        BFrag pf;
        float rs = 0.f;
        #pragma unroll
        for (int i = 0; i < 4; i++) {
            float p0 = exp2f(st[0][2 * i]     - mnew);
            float p1 = exp2f(st[0][2 * i + 1] - mnew);
            rs += p0 + p1;
            pf.u[i] = pack2bf(p0, p1);
        }
        #pragma unroll
        for (int i = 0; i < 4; i++) {
            float p0 = exp2f(st[1][2 * i]     - mnew);
            float p1 = exp2f(st[1][2 * i + 1] - mnew);
            rs += p0 + p1;
            pf.u[4 + i] = pack2bf(p0, p1);
        }
        lrun = lrun * alpha + rs;
        mrun = mnew;

        // ---- rescale accumulators: alpha broadcast to output-row form ----
        float ar[8];
        #pragma unroll
        for (int r = 0; r < 8; r++) ar[r] = __shfl(alpha, r + sel8, 32);
        #pragma unroll
        for (int s = 0; s < 4; s++)
            #pragma unroll
            for (int r = 0; r < 8; r++) oacc[s][r] *= ar[r];

        // ---- accumulate P x V^T (B column = dd, K = keys) ----
        #pragma unroll
        for (int s = 0; s < 4; s++) {
            const uint16_t* vp = vb + (size_t)(16 * s + l16) * Nc + j0 + sel8;
            BFrag vf;
            vf.q[0] = *(const uint4*)vp;
            vf.q[1] = *(const uint4*)(vp + 16);
            oacc[s] = wmma_bf16(pf, vf, oacc[s]);
        }
    }

    // ---- finalize: combine half sums, scale rows, store aT[b][n][c] ----
    float lt  = lrun + __shfl_xor(lrun, 16, 32);
    float inv = 1.0f / lt;
    float ir[8];
    #pragma unroll
    for (int r = 0; r < 8; r++) ir[r] = __shfl(inv, r + sel8, 32);

    int b = bh >> 3, head = bh & 7;
    #pragma unroll
    for (int s = 0; s < 4; s++) {
        #pragma unroll
        for (int r = 0; r < 8; r++) {
            int n = n_base + r + sel8;
            int c = head * DHEAD + 16 * s + l16;
            aT[((size_t)b * Nc + n) * Cc + c] = f2bf(oacc[s][r] * ir[r]);
        }
    }
}

// ---------------------------------------------------------------------------
// Kernel 5: proj GEMM + bias + residual.  Wave tile M=64, N=32.  f32 out.
// ---------------------------------------------------------------------------
__global__ void proj_gemm_kernel(const uint16_t* __restrict__ Pw,
                                 const uint16_t* __restrict__ aT,
                                 const float*   __restrict__ proj_b,
                                 const float*   __restrict__ x,
                                 float* __restrict__ out) {
    const int lane  = threadIdx.x & 31;
    const int l16   = lane & 15;
    const int sel8  = (lane < 16) ? 0 : 8;
    const int sel16 = (lane < 16) ? 0 : 16;

    int wid = blockIdx.x * 4 + (threadIdx.x >> 5);
    int b  = wid / 256;
    int r2 = wid % 256;
    int M0 = (r2 >> 5) * 64;
    int n0 = (r2 & 31) * 32;

    v8f acc[4][2];
    #pragma unroll
    for (int t = 0; t < 4; t++)
        #pragma unroll
        for (int s = 0; s < 2; s++)
            #pragma unroll
            for (int i = 0; i < 8; i++) acc[t][s][i] = 0.f;

    const uint16_t* aTb = aT + (size_t)b * Nc * Cc;
    for (int kb = 0; kb < Cc; kb += 32) {
        BFrag afr[4], bfr[2];
        #pragma unroll
        for (int t = 0; t < 4; t++) {
            const uint16_t* ap = Pw + (size_t)(M0 + 16 * t + l16) * Cc + kb + sel8;
            afr[t].q[0] = *(const uint4*)ap;
            afr[t].q[1] = *(const uint4*)(ap + 16);
        }
        #pragma unroll
        for (int s = 0; s < 2; s++) {
            const uint16_t* bp = aTb + (size_t)(n0 + 16 * s + l16) * Cc + kb + sel8;
            bfr[s].q[0] = *(const uint4*)bp;
            bfr[s].q[1] = *(const uint4*)(bp + 16);
        }
        #pragma unroll
        for (int t = 0; t < 4; t++)
            #pragma unroll
            for (int s = 0; s < 2; s++)
                acc[t][s] = wmma_bf16(afr[t], bfr[s], acc[t][s]);
    }

    #pragma unroll
    for (int t = 0; t < 4; t++) {
        #pragma unroll
        for (int s = 0; s < 2; s++) {
            int n = n0 + 16 * s + l16;
            #pragma unroll
            for (int r = 0; r < 8; r++) {
                int o = M0 + 16 * t + r + sel8;
                size_t idx = ((size_t)b * Cc + o) * Nc + n;
                out[idx] = acc[t][s][r] + proj_b[o] + x[idx];
            }
        }
    }
}

// ---------------------------------------------------------------------------
// Launch
// ---------------------------------------------------------------------------
extern "C" void kernel_launch(void* const* d_in, const int* in_sizes, int n_in,
                              void* d_out, int out_size, void* d_ws, size_t ws_size,
                              hipStream_t stream) {
    const float* x      = (const float*)d_in[0];
    const float* gn_w   = (const float*)d_in[1];
    const float* gn_b   = (const float*)d_in[2];
    const float* qkv_w  = (const float*)d_in[3];
    const float* qkv_b  = (const float*)d_in[4];
    const float* proj_w = (const float*)d_in[5];
    const float* proj_b = (const float*)d_in[6];
    float* out = (float*)d_out;

    char* ws = (char*)d_ws;
    uint16_t* qkv_wb  = (uint16_t*)(ws);                         // 1.5 MB
    uint16_t* proj_wb = (uint16_t*)(ws + 1572864);               // 0.5 MB
    uint16_t* hT      = (uint16_t*)(ws + 2097152);               // 16 MB ([B,N,C])
    uint16_t* qT      = (uint16_t*)(ws + 18874368);              // 16 MB ([B,h,N,d])
    uint16_t* kT      = (uint16_t*)(ws + 35651584);              // 16 MB ([B,h,N,d])
    uint16_t* vvv     = (uint16_t*)(ws + 52428800);              // 16 MB ([B,h,d,N])
    uint16_t* aT      = hT;   // hT dead after QKV GEMM: reuse for attn output

    cvt_weights_kernel<<<3072, 256, 0, stream>>>(qkv_w, proj_w, qkv_wb, proj_wb);
    groupnorm_kernel<<<Bc * 32, 256, 0, stream>>>(x, gn_w, gn_b, hT);
    qkv_gemm_kernel<<<3072, 128, 0, stream>>>(qkv_wb, hT, qkv_b, qT, kT, vvv);
    attn_kernel<<<1024, 256, 0, stream>>>(qT, kT, vvv, aT);
    proj_gemm_kernel<<<1024, 128, 0, stream>>>(proj_wb, aT, proj_b, x, out);
}
